// MultiHeadDepthwiseSelfAttention_23038204576347
// MI455X (gfx1250) — matrix operand order repaired and checked by reading, not compiled, not run on hardware
//
#include <hip/hip_runtime.h>
#include <hip/hip_bf16.h>
#include <math.h>

// ---------------------------------------------------------------------------
// Multi-head depthwise self-attention, flash-attention style, CDNA5 wave32.
//   B=16, N=1024, F=768, H=12, d=64
//   q = (x*wq+bq)/sqrt(F);  S = q K^T;  P = softmax(S);  O = P V;  out = O*wo+bo
//
// Fast path (needs 50.3MB workspace):
//   prepass 1: k16[B][N][F]      = f16(x*wk+bk)      (natural layout)
//   prepass 2: v16T[B][H][64][N] = f16(x*wv+bv)^T    (LDS-tiled transpose)
//   main: one wave per 16-query tile, 64 keys/step:
//     - 8 wmma for S = Q K^T (b128 f16 B-fragment loads)
//     - row-max via f16-packed xor butterflies (16 ds_bpermute / 64 keys)
//     - row-sum accumulated BY THE TENSOR CORE via an all-ones B fragment
//       (every lane of Osum holds ell[r]; zero shuffles)
//     - 10 wmma for O += P V (+ ones column), transposed f16 V loads
// Fallback path: round-1 kernel (f32 loads + in-loop conversion).
// ---------------------------------------------------------------------------

typedef _Float16 v16h __attribute__((ext_vector_type(16)));
typedef _Float16 v8h  __attribute__((ext_vector_type(8)));
typedef _Float16 v4h  __attribute__((ext_vector_type(4)));
typedef _Float16 v2h  __attribute__((ext_vector_type(2)));
typedef float    v8f  __attribute__((ext_vector_type(8)));
typedef float    v4f  __attribute__((ext_vector_type(4)));

#define BATCH 16
#define SEQ   1024
#define FEAT  768
#define NHEAD 12
#define DHEAD 64

__device__ __forceinline__ v8f wmma16(v16h a, v16h b, v8f c) {
  // D = A(16x32, f16) * B(32x16, f16) + C(16x16, f32)
  return __builtin_amdgcn_wmma_f32_16x16x32_f16(
      /*neg_a=*/false, a, /*neg_b=*/false, b,
      /*c_mod=*/(short)0, c, /*reuse_a=*/false, /*reuse_b=*/false);
}

// ---------------------------------------------------------------------------
// Pre-pass 1: k16 = f16(x*wk + bk), natural [B][N][F] layout.
// ---------------------------------------------------------------------------
__global__ __launch_bounds__(256) void kcast_kernel(
    const float* __restrict__ x,
    const float* __restrict__ wk, const float* __restrict__ bk,
    _Float16* __restrict__ k16)
{
  const size_t i4   = (size_t)blockIdx.x * 256 + threadIdx.x;  // one float4
  const size_t base = i4 * 4;
  const int    c    = (int)(base % FEAT);
  v4f v = *(const v4f*)(x + base);
  v4h o;
  #pragma unroll
  for (int j = 0; j < 4; ++j)
    o[j] = (_Float16)(v[j] * wk[c + j] + bk[c + j]);
  *(v4h*)(k16 + base) = o;
}

// ---------------------------------------------------------------------------
// Pre-pass 2: v16T[b][h][c][n] = f16(x[b][n][h*64+c]*wv + bv).
// 64x64 tile transpose through LDS; both global sides coalesced.
// ---------------------------------------------------------------------------
__global__ __launch_bounds__(256) void vtrans_kernel(
    const float* __restrict__ x,
    const float* __restrict__ wv, const float* __restrict__ bv,
    _Float16* __restrict__ v16T)
{
  __shared__ __align__(16) _Float16 tileT[64][72];   // [c][n], padded rows

  const int tid = threadIdx.x;
  const int blk = blockIdx.x;            // (b*NHEAD + h)*16 + nt
  const int nt  = blk & 15;
  const int h   = (blk >> 4) % NHEAD;
  const int b   = blk / (16 * NHEAD);
  const int n0  = nt * 64;
  const int ch0 = h * DHEAD;

  const int nr = tid >> 4;               // 0..15
  const int c4 = (tid & 15) * 4;
  #pragma unroll
  for (int pass = 0; pass < 4; ++pass) {
    const int n = nr + pass * 16;
    v4f v = *(const v4f*)(x + ((size_t)(b * SEQ + n0 + n)) * FEAT + ch0 + c4);
    #pragma unroll
    for (int j = 0; j < 4; ++j)
      tileT[c4 + j][n] = (_Float16)(v[j] * wv[ch0 + c4 + j] + bv[ch0 + c4 + j]);
  }
  __syncthreads();

  const int cr = tid >> 3;               // 0..31
  const int n8 = (tid & 7) * 8;
  #pragma unroll
  for (int pass = 0; pass < 2; ++pass) {
    const int c = cr + pass * 32;
    v8h o = *(const v8h*)(&tileT[c][n8]);
    *(v8h*)(v16T + (((size_t)(b * NHEAD + h) * DHEAD + c) * SEQ + n0 + n8)) = o;
  }
}

// ---------------------------------------------------------------------------
// Fast main kernel: f16 K/V from workspace, 64 keys per step.
// ---------------------------------------------------------------------------
__global__ __launch_bounds__(256) void mhdsa_flash_f16kv(
    const float* __restrict__ x,
    const float* __restrict__ wq, const float* __restrict__ bq,
    const _Float16* __restrict__ k16,
    const _Float16* __restrict__ v16T,
    const float* __restrict__ wo, const float* __restrict__ bo,
    float* __restrict__ out)
{
  __shared__ __align__(16) _Float16 ldsP[8][16 * 72];  // 16 rows x 64 keys, pad 72

  const int lane = threadIdx.x & 31;
  const int wave = threadIdx.x >> 5;
  const int tile = blockIdx.x * 8 + wave;        // 0 .. 12287
  const int qt   = tile & 63;
  const int h    = (tile >> 6) % NHEAD;
  const int b    = tile / (64 * NHEAD);

  const int nl   = lane & 15;
  const int hi   = lane >> 4;
  const int h8   = hi * 8;
  const int hi16 = hi * 16;
  const int q0   = qt * 16;
  const int ch0  = h * DHEAD;

  const float qscale = 0.03608439182435161f;     // 1/sqrt(768)

  // ---- Q tile -> two A fragments (amortized over whole key loop).
  v16h aQ[2];
  {
    const float* xr = x + ((size_t)(b * SEQ + q0 + nl)) * FEAT + ch0;
    #pragma unroll
    for (int half = 0; half < 2; ++half) {
      #pragma unroll
      for (int chunk = 0; chunk < 2; ++chunk) {
        const int c0 = half * 32 + chunk * 16 + h8;
        v4f lo = *(const v4f*)(xr + c0);
        v4f hv = *(const v4f*)(xr + c0 + 4);
        #pragma unroll
        for (int j = 0; j < 4; ++j) {
          aQ[half][chunk * 8 + j] =
              (_Float16)((lo[j] * wq[ch0 + c0 + j] + bq[ch0 + c0 + j]) * qscale);
          aQ[half][chunk * 8 + 4 + j] =
              (_Float16)((hv[j] * wq[ch0 + c0 + 4 + j] + bq[ch0 + c0 + 4 + j]) * qscale);
        }
      }
    }
  }

  // All-ones B fragment: every column of wmma(P, ones) is the row-sum of P.
  v16h bOnes;
  #pragma unroll
  for (int j = 0; j < 16; ++j) bOnes[j] = (_Float16)1.0f;

  const v8f vzero = {0.f, 0.f, 0.f, 0.f, 0.f, 0.f, 0.f, 0.f};
  v8f   Oacc[4];
  v8f   Osum = vzero;            // running ell[r], replicated across lanes
  float mrow[8];
  #pragma unroll
  for (int t = 0; t < 4; ++t) Oacc[t] = vzero;
  #pragma unroll
  for (int r = 0; r < 8; ++r) mrow[r] = -INFINITY;

  _Float16* Pb = &ldsP[wave][0];
  const _Float16* vbase = v16T + ((size_t)(b * NHEAD + h) * DHEAD) * SEQ;

  for (int j0 = 0; j0 < SEQ; j0 += 64) {
    if (j0 + 64 < SEQ) {   // uniform; emits global_prefetch_b8
      // next 64 K rows: each head-row slice (64 f16 = 128B) is one line
      __builtin_prefetch(k16 + ((size_t)(b * SEQ + j0 + 64 + lane * 2)) * FEAT + ch0, 0, 1);
      // next 64 keys of all 64 V channels (128B per channel)
      __builtin_prefetch(vbase + (size_t)(lane * 2) * SEQ + j0 + 64, 0, 1);
      __builtin_prefetch(vbase + (size_t)(lane * 2 + 1) * SEQ + j0 + 64, 0, 1);
    }

    // ---- S = Q K^T : 4 key-tiles of 16, B fragments = two b128 f16 loads.
    v8f S[4];
    #pragma unroll
    for (int s = 0; s < 4; ++s) {
      const _Float16* kr = k16 + ((size_t)(b * SEQ + j0 + s * 16 + nl)) * FEAT + ch0;
      v8f acc = vzero;
      #pragma unroll
      for (int half = 0; half < 2; ++half) {
        const int c0 = half * 32 + h8;
        v8h lo = *(const v8h*)(kr + c0);
        v8h hv = *(const v8h*)(kr + c0 + 16);
        v16h bK;
        #pragma unroll
        for (int j = 0; j < 8; ++j) { bK[j] = lo[j]; bK[8 + j] = hv[j]; }
        acc = wmma16(aQ[half], bK, acc);
      }
      S[s] = acc;
    }

    // ---- Block row-max: pack row pairs into f16x2, xor-butterfly (16 lanes).
    // f16-rounded max is still a valid softmax stabilizer (<=1 ulp off).
    float mblk[8];
    #pragma unroll
    for (int r = 0; r < 8; ++r)
      mblk[r] = fmaxf(fmaxf(S[0][r], S[1][r]), fmaxf(S[2][r], S[3][r]));
    #pragma unroll
    for (int pr = 0; pr < 4; ++pr) {
      v2h hm;
      hm[0] = (_Float16)mblk[2 * pr];
      hm[1] = (_Float16)mblk[2 * pr + 1];
      #pragma unroll
      for (int mask = 1; mask <= 8; mask <<= 1) {
        int o = __shfl_xor(__builtin_bit_cast(int, hm), mask, 32);
        v2h ov = __builtin_bit_cast(v2h, o);
        hm[0] = hm[0] > ov[0] ? hm[0] : ov[0];
        hm[1] = hm[1] > ov[1] ? hm[1] : ov[1];
      }
      mblk[2 * pr]     = (float)hm[0];
      mblk[2 * pr + 1] = (float)hm[1];
    }

    // ---- Online update: rescale accumulators, write P (f16) to LDS.
    #pragma unroll
    for (int r = 0; r < 8; ++r) {
      const float mn    = fmaxf(mrow[r], mblk[r]);
      const float alpha = __expf(mrow[r] - mn);
      mrow[r] = mn;
      #pragma unroll
      for (int t = 0; t < 4; ++t) Oacc[t][r] *= alpha;
      Osum[r] *= alpha;
      const int row = r + h8;
      Pb[row * 72 + nl]      = (_Float16)__expf(S[0][r] - mn);
      Pb[row * 72 + 16 + nl] = (_Float16)__expf(S[1][r] - mn);
      Pb[row * 72 + 32 + nl] = (_Float16)__expf(S[2][r] - mn);
      Pb[row * 72 + 48 + nl] = (_Float16)__expf(S[3][r] - mn);
    }

    asm volatile("s_wait_dscnt 0" ::: "memory");

    // ---- P (16x64) back as two A fragments (b128 LDS loads).
    v16h aP[2];
    #pragma unroll
    for (int f = 0; f < 2; ++f) {
      v8h c0 = *(const v8h*)(Pb + nl * 72 + f * 32 + h8);
      v8h c1 = *(const v8h*)(Pb + nl * 72 + f * 32 + 16 + h8);
      #pragma unroll
      for (int j = 0; j < 8; ++j) { aP[f][j] = c0[j]; aP[f][8 + j] = c1[j]; }
    }

    // ---- O += P V (transposed f16 V), Osum += P * ones (row-sums).
    #pragma unroll
    for (int f = 0; f < 2; ++f) {
      #pragma unroll
      for (int t = 0; t < 4; ++t) {
        const int c = t * 16 + nl;
        const _Float16* pv = vbase + (size_t)c * SEQ + j0 + f * 32 + h8;
        v8h lo = *(const v8h*)(pv);
        v8h hv = *(const v8h*)(pv + 16);
        v16h bV;
        #pragma unroll
        for (int j = 0; j < 8; ++j) { bV[j] = lo[j]; bV[8 + j] = hv[j]; }
        Oacc[t] = wmma16(aP[f], bV, Oacc[t]);
      }
      Osum = wmma16(aP[f], bOnes, Osum);
    }

    asm volatile("s_wait_dscnt 0" ::: "memory");  // WAR guard on Pb
  }

  // ---- Epilogue: normalize + depthwise output projection.
  #pragma unroll
  for (int r = 0; r < 8; ++r) {
    const float  inv     = 1.0f / Osum[r];
    const size_t rowbase = ((size_t)(b * SEQ + q0 + r + h8)) * FEAT + ch0;
    #pragma unroll
    for (int t = 0; t < 4; ++t) {
      const int c = t * 16 + nl;
      out[rowbase + c] = Oacc[t][r] * inv * wo[ch0 + c] + bo[ch0 + c];
    }
  }
}

// ---------------------------------------------------------------------------
// Fallback main kernel (round-1 version): no workspace required.
// ---------------------------------------------------------------------------
__global__ __launch_bounds__(256) void mhdsa_flash_fallback(
    const float* __restrict__ x,
    const float* __restrict__ wq, const float* __restrict__ bq,
    const float* __restrict__ wk, const float* __restrict__ bk,
    const float* __restrict__ wv, const float* __restrict__ bv,
    const float* __restrict__ wo, const float* __restrict__ bo,
    float* __restrict__ out)
{
  __shared__ __align__(16) _Float16 ldsP[8][16 * 40];

  const int lane = threadIdx.x & 31;
  const int wave = threadIdx.x >> 5;
  const int tile = blockIdx.x * 8 + wave;
  const int qt   = tile & 63;
  const int h    = (tile >> 6) % NHEAD;
  const int b    = tile / (64 * NHEAD);

  const int nl   = lane & 15;
  const int hi   = lane >> 4;
  const int h8   = hi * 8;
  const int hi16 = hi * 16;
  const int q0   = qt * 16;
  const int ch0  = h * DHEAD;

  const float qscale = 0.03608439182435161f;

  v16h aQ[2];
  {
    const float* xr = x + ((size_t)(b * SEQ + q0 + nl)) * FEAT + ch0;
    #pragma unroll
    for (int half = 0; half < 2; ++half) {
      #pragma unroll
      for (int chunk = 0; chunk < 2; ++chunk) {
        const int c0 = half * 32 + chunk * 16 + h8;
        v4f lo = *(const v4f*)(xr + c0);
        v4f hv = *(const v4f*)(xr + c0 + 4);
        #pragma unroll
        for (int j = 0; j < 4; ++j) {
          aQ[half][chunk * 8 + j] =
              (_Float16)((lo[j] * wq[ch0 + c0 + j] + bq[ch0 + c0 + j]) * qscale);
          aQ[half][chunk * 8 + 4 + j] =
              (_Float16)((hv[j] * wq[ch0 + c0 + 4 + j] + bq[ch0 + c0 + 4 + j]) * qscale);
        }
      }
    }
  }

  const v8f vzero = {0.f, 0.f, 0.f, 0.f, 0.f, 0.f, 0.f, 0.f};
  v8f   Oacc[4];
  float mrow[8], lrow[8];
  #pragma unroll
  for (int t = 0; t < 4; ++t) Oacc[t] = vzero;
  #pragma unroll
  for (int r = 0; r < 8; ++r) { mrow[r] = -INFINITY; lrow[r] = 0.0f; }

  _Float16* Pb = &ldsP[wave][0];

  for (int j0 = 0; j0 < SEQ; j0 += 32) {
    v8f S[2];
    #pragma unroll
    for (int s = 0; s < 2; ++s) {
      const float* kr = x + ((size_t)(b * SEQ + j0 + s * 16 + nl)) * FEAT + ch0;
      v8f acc = vzero;
      #pragma unroll
      for (int half = 0; half < 2; ++half) {
        const int c0 = half * 32 + h8;
        v16h bK;
        #pragma unroll
        for (int chunk = 0; chunk < 4; ++chunk) {
          v4f v = *(const v4f*)(kr + c0 + chunk * 4 + (chunk >> 1) * 8);
          #pragma unroll
          for (int j = 0; j < 4; ++j) {
            const int c = c0 + chunk * 4 + (chunk >> 1) * 8 + j;
            bK[chunk * 4 + j] = (_Float16)(v[j] * wk[ch0 + c] + bk[ch0 + c]);
          }
        }
        acc = wmma16(aQ[half], bK, acc);
      }
      S[s] = acc;
    }

    #pragma unroll
    for (int r = 0; r < 8; ++r) {
      float s0 = S[0][r], s1 = S[1][r];
      float rm = fmaxf(s0, s1);
      #pragma unroll
      for (int mask = 1; mask <= 8; mask <<= 1)
        rm = fmaxf(rm, __shfl_xor(rm, mask, 32));
      const float mn    = fmaxf(mrow[r], rm);
      const float alpha = __expf(mrow[r] - mn);
      const float p0    = __expf(s0 - mn);
      const float p1    = __expf(s1 - mn);
      float rs = p0 + p1;
      #pragma unroll
      for (int mask = 1; mask <= 8; mask <<= 1)
        rs += __shfl_xor(rs, mask, 32);
      lrow[r] = lrow[r] * alpha + rs;
      mrow[r] = mn;
      #pragma unroll
      for (int t = 0; t < 4; ++t) Oacc[t][r] *= alpha;
      const int row = r + h8;
      Pb[row * 40 + nl]      = (_Float16)p0;
      Pb[row * 40 + 16 + nl] = (_Float16)p1;
    }

    asm volatile("s_wait_dscnt 0" ::: "memory");

    v16h aP;
    {
      v8h c0 = *(const v8h*)(Pb + nl * 40 + h8);
      v8h c1 = *(const v8h*)(Pb + nl * 40 + 16 + h8);
      #pragma unroll
      for (int j = 0; j < 8; ++j) { aP[j] = c0[j]; aP[8 + j] = c1[j]; }
    }

    #pragma unroll
    for (int t = 0; t < 4; ++t) {
      const int   c   = t * 16 + nl;
      const float wvc = wv[ch0 + c];
      const float bvc = bv[ch0 + c];
      const float* pv = x + ((size_t)(b * SEQ + j0 + h8)) * FEAT + ch0 + c;
      v16h bV;
      #pragma unroll
      for (int i = 0; i < 16; ++i)
        bV[i] = (_Float16)(pv[(size_t)(i + (i & 8)) * FEAT] * wvc + bvc);
      Oacc[t] = wmma16(aP, bV, Oacc[t]);
    }

    asm volatile("s_wait_dscnt 0" ::: "memory");
  }

  #pragma unroll
  for (int r = 0; r < 8; ++r) {
    const float  inv     = 1.0f / lrow[r];
    const size_t rowbase = ((size_t)(b * SEQ + q0 + r + h8)) * FEAT + ch0;
    #pragma unroll
    for (int t = 0; t < 4; ++t) {
      const int c = t * 16 + nl;
      out[rowbase + c] = Oacc[t][r] * inv * wo[ch0 + c] + bo[ch0 + c];
    }
  }
}

extern "C" void kernel_launch(void* const* d_in, const int* in_sizes, int n_in,
                              void* d_out, int out_size, void* d_ws, size_t ws_size,
                              hipStream_t stream) {
  const float* x  = (const float*)d_in[0];
  const float* wq = (const float*)d_in[1];
  const float* bq = (const float*)d_in[2];
  const float* wk = (const float*)d_in[3];
  const float* bk = (const float*)d_in[4];
  const float* wv = (const float*)d_in[5];
  const float* bv = (const float*)d_in[6];
  const float* wo = (const float*)d_in[7];
  const float* bo = (const float*)d_in[8];
  float* out = (float*)d_out;

  const size_t elems = (size_t)BATCH * SEQ * FEAT;       // 12,582,912
  const size_t need  = 2 * elems * sizeof(_Float16);     // 50,331,648 B
  const int tiles    = BATCH * NHEAD * (SEQ / 16);       // 12288 wave-tiles
  const int blocks   = tiles / 8;

  if (ws_size >= need) {
    _Float16* k16  = (_Float16*)d_ws;
    _Float16* v16T = k16 + elems;
    hipLaunchKernelGGL(kcast_kernel, dim3((int)(elems / 4 / 256)), dim3(256),
                       0, stream, x, wk, bk, k16);
    hipLaunchKernelGGL(vtrans_kernel, dim3(BATCH * NHEAD * (SEQ / 64)), dim3(256),
                       0, stream, x, wv, bv, v16T);
    hipLaunchKernelGGL(mhdsa_flash_f16kv, dim3(blocks), dim3(256), 0, stream,
                       x, wq, bq, k16, v16T, wo, bo, out);
  } else {
    hipLaunchKernelGGL(mhdsa_flash_fallback, dim3(blocks), dim3(256), 0, stream,
                       x, wq, bq, wk, bk, wv, bv, wo, bo, out);
  }
}
